// _WindowAttention_25718264168506
// MI455X (gfx1250) — matrix operand order repaired and checked by reading, not compiled, not run on hardware
//
#include <hip/hip_runtime.h>

// ---------------------------------------------------------------------------
// Fused Swin window-attention for gfx1250 (MI455X), wave32 + WMMA.
// Two kernels per launch:
//   1) prep: qkv_w/proj_w f32 -> f16 into d_ws (shared by all 4096 blocks,
//      halves weight VMEM bytes and removes all per-block cvt VALU).
//   2) main: per block (one window b), 512 threads = 16 wave32 waves:
//        x[49,128] -> LDS f16
//        qkv  = x @ qkv_w^T + qkv_b       (WMMA f16->f32), q scaled, v^T
//        s    = q @ k^T  (per head, hd=32 -> single WMMA K-step)
//        s   += rel_pos_bias + mask ; softmax (f32 LDS, in-place f32->f16)
//        ctx  = softmax(s) @ v            (WMMA)
//        out  = ctx @ proj_w^T + proj_b   (WMMA, f32 out to HBM)
// Static wave->tile assignment: uniform control flow (EXEC all-1s around
// every WMMA), A-fragments hoisted and reused across N-tiles.
// ---------------------------------------------------------------------------

typedef __attribute__((ext_vector_type(16))) _Float16 v16h;
typedef __attribute__((ext_vector_type(8)))  float    v8f;
typedef _Float16 h16;

#define N_TOK 49
#define NPAD  64
#define CDIM  128
#define NH_   4
#define HD    32

#define SQK   136   // q/k/x/ctx row stride (halves)
#define SVT   72    // v^T row stride (halves)
#define SSC   68    // scores row stride (floats); as halves: 136

// LDS map (bytes):               size
#define OFF_Q   0                 // 64*136*2 = 17408
#define OFF_K   17408             // 17408
#define OFF_VT  34816             // 128*72*2 = 18432
#define OFF_SC  53248             // 4*64*68*4 = 69632
#define OFF_X   122880            // 17408 (x, reused as ctx)
#define SMEM_SZ 140288

#define QKVW_ELEMS (3 * CDIM * CDIM)   // 49152
#define PROJW_ELEMS (CDIM * CDIM)      // 16384

#define WMMA16(a,b,c) __builtin_amdgcn_wmma_f32_16x16x32_f16( \
    false, (a), false, (b), (short)0, (c), false, false)

// A fragment: 16x32 f16, row-major source, lane m = L%16, k = 16*(L/16) + e (contiguous, matches B)
__device__ __forceinline__ v16h lds_load_a16(const h16* base, int stride, int lane) {
  v16h a;
  const h16* row = base + (lane & 15) * stride;
  const int khi = (lane & 16) ? 16 : 0;
#pragma unroll
  for (int v = 0; v < 8; ++v) {
    const int k = khi + (v << 1);
    a[2 * v]     = row[k];
    a[2 * v + 1] = row[k + 1];
  }
  return a;
}

// B fragment: element e at p[e] (caller bakes in n = L%16 and K half-select);
// 16 contiguous halves -> compiler merges into 2x b128 loads.
__device__ __forceinline__ v16h load_b16(const h16* p) {
  v16h b;
#pragma unroll
  for (int e = 0; e < 16; ++e) b[e] = p[e];
  return b;
}

// ---- Kernel 1: weight f32 -> f16 conversion into workspace ----------------
__global__ __launch_bounds__(256)
void swin_weights_to_f16(const float* __restrict__ qkv_w,
                         const float* __restrict__ proj_w,
                         h16* __restrict__ wh) {
  const int i = blockIdx.x * 256 + threadIdx.x;
  if (i < QKVW_ELEMS)
    wh[i] = (h16)qkv_w[i];
  else if (i < QKVW_ELEMS + PROJW_ELEMS)
    wh[i] = (h16)proj_w[i - QKVW_ELEMS];
}

// ---- Kernel 2: fused window attention --------------------------------------
__global__ __launch_bounds__(512, 1)
void swin_window_attn_gfx1250(const float* __restrict__ x,
                              const float* __restrict__ mask,
                              const h16*   __restrict__ wh,     // f16 weights
                              const float* __restrict__ qkv_b,
                              const float* __restrict__ proj_b,
                              const float* __restrict__ bias_table,
                              float* __restrict__ out, int nW) {
  __shared__ __align__(16) unsigned char smem[SMEM_SZ];
  h16*   qh  = (h16*)(smem + OFF_Q);    // [64][136] f16 (pre-scaled)
  h16*   kh  = (h16*)(smem + OFF_K);    // [64][136] f16
  h16*   vT  = (h16*)(smem + OFF_VT);   // [128][72] f16 (transposed)
  float* sc  = (float*)(smem + OFF_SC); // [4][64][68] f32, reused as f16 attn
  h16*   xh  = (h16*)(smem + OFF_X);    // [64][136] f16; reused as ctx
  h16*   ctx = xh;

  const h16* qkv_wh  = wh;                // [384][128] f16
  const h16* proj_wh = wh + QKVW_ELEMS;   // [128][128] f16

  const int   b      = blockIdx.x;
  const int   tid    = threadIdx.x;
  const int   lane   = tid & 31;
  const int   wave   = tid >> 5;              // 0..15
  const int   lanen  = lane & 15;
  const int   hi     = (lane & 16) ? 8 : 0;   // C/D row offset
  const int   klo    = (lane & 16) ? 16 : 0;  // B K half-select
  const float scale  = 0.17677669529663687f;  // 1/sqrt(32)

  // ---- Phase 0: x -> LDS f16 (pure copy), then zero-pad rows 49..63 --------
  const float* xb = x + (size_t)b * N_TOK * CDIM;
  for (int idx = tid; idx < N_TOK * CDIM; idx += 512)
    xh[(idx >> 7) * SQK + (idx & 127)] = (h16)xb[idx];
  for (int idx = N_TOK * CDIM + tid; idx < NPAD * CDIM; idx += 512)
    xh[(idx >> 7) * SQK + (idx & 127)] = (h16)0.0f;
  __syncthreads();

  // ---- Phase 1: qkv = x @ qkv_w^T + qkv_b  (M=64, N=384, K=128) ------------
  // wave -> (mt, jgrp): A-frags for one M-tile loaded once, 6 N-tiles each.
  {
    const int mt = wave >> 2, jgrp = wave & 3;
    v16h a[4];
#pragma unroll
    for (int kk = 0; kk < 4; ++kk)
      a[kk] = lds_load_a16(xh + mt * 16 * SQK + kk * 32, SQK, lane);

#pragma unroll
    for (int jt = jgrp; jt < 24; jt += 4) {   // 6 iterations, uniform per wave
      v8f acc = {};
#pragma unroll
      for (int kk = 0; kk < 4; ++kk) {
        v16h bfr = load_b16(qkv_wh + (jt * 16 + lanen) * CDIM + kk * 32 + klo);
        acc = WMMA16(a[kk], bfr, acc);
      }
      // q/k/v routing is uniform per wave (jt-tile aligned to the 128-splits)
      if (jt < 8) {                            // -> q (pre-scaled)
        const int   j  = jt * 16 + lanen;
        const float bj = qkv_b[j];
#pragma unroll
        for (int r = 0; r < 8; ++r)
          qh[(mt * 16 + r + hi) * SQK + j] = (h16)((acc[r] + bj) * scale);
      } else if (jt < 16) {                    // -> k
        const int   j  = (jt - 8) * 16 + lanen;
        const float bj = qkv_b[128 + j];
#pragma unroll
        for (int r = 0; r < 8; ++r)
          kh[(mt * 16 + r + hi) * SQK + j] = (h16)(acc[r] + bj);
      } else {                                 // -> v, transposed
        const int   j  = (jt - 16) * 16 + lanen;
        const float bj = qkv_b[256 + j];
#pragma unroll
        for (int r = 0; r < 8; ++r)
          vT[j * SVT + (mt * 16 + r + hi)] = (h16)(acc[r] + bj);
      }
    }
  }
  __syncthreads();

  // ---- Phase 2: scores = q @ k^T per head (hd=32 -> 1 WMMA K-step) ---------
  // wave -> (h, mt): one A-frag reused across 4 N-tiles.
  {
    const int h = wave >> 2, mt = wave & 3;
    v16h a = lds_load_a16(qh + mt * 16 * SQK + h * HD, SQK, lane);
    float* sh = sc + h * (NPAD * SSC);
#pragma unroll
    for (int nt = 0; nt < 4; ++nt) {
      v16h bfr = load_b16(kh + (nt * 16 + lanen) * SQK + h * HD + klo);
      v8f  acc = {};
      acc = WMMA16(a, bfr, acc);
#pragma unroll
      for (int r = 0; r < 8; ++r)
        sh[(mt * 16 + r + hi) * SSC + nt * 16 + lanen] = acc[r];
    }
  }
  __syncthreads();

  // ---- Phase 3: + rel-pos bias + mask, softmax; compact f32->f16 in place --
  if (tid < NH_ * NPAD) {
    const int h = tid >> 6, row = tid & 63;
    float*          srow = sc + h * (NPAD * SSC) + row * SSC;
    volatile float* vrow = srow;
    volatile h16*   arow = (volatile h16*)srow;  // f16 attn aliases f32 row
    if (row < N_TOK) {
      const float* mrow = mask + (size_t)(b % nW) * N_TOK * N_TOK + row * N_TOK;
      const int r0 = row / 7, r1 = row - r0 * 7;
      float mx = -3.0e38f;
      int col = 0;
      for (int c0 = 0; c0 < 7; ++c0)
        for (int c1 = 0; c1 < 7; ++c1, ++col) {
          const int ridx = (r0 - c0 + 6) * 13 + (r1 - c1 + 6);
          const float s = srow[col] + bias_table[ridx * NH_ + h] + mrow[col];
          srow[col] = s;
          mx = fmaxf(mx, s);
        }
      float sum = 0.0f;
      for (int c = 0; c < N_TOK; ++c) sum += __expf(srow[c] - mx);
      const float inv = 1.0f / sum;
      for (int c = 0; c < NPAD; ++c) {  // read f32 col c, then write f16 col c
        const float e = (c < N_TOK) ? __expf(vrow[c] - mx) * inv : 0.0f;
        arow[c] = (h16)e;
      }
    } else {
      for (int c = 0; c < NPAD; ++c) arow[c] = (h16)0.0f;  // padded query rows
    }
  }
  __syncthreads();

  // ---- Phase 4: ctx = attn @ v  (M=64, N=32/head, K=64 -> 2 K-steps) -------
  // wave -> (h, mt): both K-step A-frags hoisted, reused across 2 N-tiles.
  {
    const int h = wave >> 2, mt = wave & 3;
    const h16* attn = (const h16*)sc;  // head stride NPAD*SSC*2 halves, row 136
    const h16* abase = attn + h * (NPAD * SSC * 2) + mt * 16 * (SSC * 2);
    v16h a[2];
#pragma unroll
    for (int kk = 0; kk < 2; ++kk)
      a[kk] = lds_load_a16(abase + kk * 32, SSC * 2, lane);

#pragma unroll
    for (int dt = 0; dt < 2; ++dt) {
      v8f acc = {};
#pragma unroll
      for (int kk = 0; kk < 2; ++kk) {
        v16h bfr = load_b16(vT + (h * HD + dt * 16 + lanen) * SVT + kk * 32 + klo);
        acc = WMMA16(a[kk], bfr, acc);
      }
#pragma unroll
      for (int r = 0; r < 8; ++r)
        ctx[(mt * 16 + r + hi) * SQK + h * HD + dt * 16 + lanen] = (h16)acc[r];
    }
  }
  __syncthreads();

  // ---- Phase 5: out = ctx @ proj_w^T + proj_b  (M=49, N=128, K=128) --------
  // wave -> (mt, jgrp): 4 A-frags hoisted, 2 N-tiles each.
  {
    const int mt = wave >> 2, jgrp = wave & 3;
    float* ob = out + (size_t)b * N_TOK * CDIM;
    v16h a[4];
#pragma unroll
    for (int kk = 0; kk < 4; ++kk)
      a[kk] = lds_load_a16(ctx + mt * 16 * SQK + kk * 32, SQK, lane);

#pragma unroll
    for (int jt = jgrp; jt < 8; jt += 4) {    // 2 iterations, uniform per wave
      v8f acc = {};
#pragma unroll
      for (int kk = 0; kk < 4; ++kk) {
        v16h bfr = load_b16(proj_wh + (jt * 16 + lanen) * CDIM + kk * 32 + klo);
        acc = WMMA16(a[kk], bfr, acc);
      }
      const int   j  = jt * 16 + lanen;
      const float pb = proj_b[j];
#pragma unroll
      for (int r = 0; r < 8; ++r) {
        const int m = mt * 16 + r + hi;
        if (m < N_TOK) ob[m * CDIM + j] = acc[r] + pb;
      }
    }
  }
}

extern "C" void kernel_launch(void* const* d_in, const int* in_sizes, int n_in,
                              void* d_out, int out_size, void* d_ws, size_t ws_size,
                              hipStream_t stream) {
  const float* x          = (const float*)d_in[0];
  const float* mask       = (const float*)d_in[1];
  const float* qkv_w      = (const float*)d_in[2];
  const float* qkv_b      = (const float*)d_in[3];
  const float* proj_w     = (const float*)d_in[4];
  const float* proj_b     = (const float*)d_in[5];
  const float* bias_table = (const float*)d_in[6];
  float* out = (float*)d_out;
  h16*   wh  = (h16*)d_ws;   // 65536 halves = 128 KB

  const int B_ = in_sizes[0] / (N_TOK * CDIM);        // 4096 windows
  const int nW = in_sizes[1] / (N_TOK * N_TOK);       // 64 masks

  const int wtot = QKVW_ELEMS + PROJW_ELEMS;          // 65536
  hipLaunchKernelGGL(swin_weights_to_f16, dim3((wtot + 255) / 256), dim3(256),
                     0, stream, qkv_w, proj_w, wh);

  hipLaunchKernelGGL(swin_window_attn_gfx1250, dim3(B_), dim3(512), 0, stream,
                     x, mask, wh, qkv_b, proj_b, bias_table, out, nW);
}
